// equivariant_decoder_5463198401070
// MI455X (gfx1250) — matrix operand order repaired and checked by reading, not compiled, not run on hardware
//
#include <hip/hip_runtime.h>
#include <hip/hip_bf16.h>

// ---------------------------------------------------------------------------
// N=1024, C=256, P=512, F=768, Q=5120; out pitch = 3*P + Q = 6656
// Dominant work: t = r_dec @ Wtp (1024 x 256 x 131072 bf16 GEMM, 68.7 GFLOP)
// fused in-register with out[b,p,m] = (1/C) * sum_j t[b,j,p] * v[b,j,m].
// Wtp (67 MB bf16) resides in the 192 MB L2 -> compute-bound on WMMA.
// ---------------------------------------------------------------------------

typedef __bf16 bf16_t;
typedef bf16_t v16bf __attribute__((ext_vector_type(16)));
typedef bf16_t v8bf  __attribute__((ext_vector_type(8)));
typedef float  v8f   __attribute__((ext_vector_type(8)));

// ---- WMMA fragment loaders (contiguous k = 16*half + s, same order for A and B) --------------
__device__ __forceinline__ v16bf load_afrag(const bf16_t* p) {
  v8bf lo = *(const v8bf*)p;          // K = base .. base+7
  v8bf hi = *(const v8bf*)(p + 8);    // K = base+8 .. base+15
  v16bf f;
#pragma unroll
  for (int e = 0; e < 8; ++e) { f[e] = lo[e]; f[e + 8] = hi[e]; }
  return f;
}
__device__ __forceinline__ v16bf load_bfrag(const bf16_t* p) {
  v8bf lo = *(const v8bf*)p;
  v8bf hi = *(const v8bf*)(p + 8);
  v16bf f;
#pragma unroll
  for (int e = 0; e < 8; ++e) { f[e] = lo[e]; f[e + 8] = hi[e]; }
  return f;
}

// ---- tiled transpose + f32->bf16 convert: out[c*R + r] = in[r*Cc + c] ------
__global__ void k_transpose_cvt(const float* __restrict__ in,
                                bf16_t* __restrict__ out, int R, int Cc) {
  __shared__ float tile[32][33];
  int r0 = blockIdx.x * 32, c0 = blockIdx.y * 32;
  int tx = threadIdx.x, ty = threadIdx.y;   // 32 x 8
#pragma unroll
  for (int k = 0; k < 4; ++k)
    tile[ty + 8 * k][tx] = in[(size_t)(r0 + ty + 8 * k) * Cc + c0 + tx];
  __syncthreads();
#pragma unroll
  for (int k = 0; k < 4; ++k)
    out[(size_t)(c0 + ty + 8 * k) * R + r0 + tx] = (bf16_t)tile[tx][ty + 8 * k];
}

// ---- r = ||v|| -------------------------------------------------------------
__global__ void k_norm(const float* __restrict__ enc, bf16_t* __restrict__ r) {
  int idx = blockIdx.x * blockDim.x + threadIdx.x;   // N*C threads
  int b = idx >> 8, c = idx & 255;
  const float* p = enc + (size_t)b * 768 + c * 3;
  float x = p[0], y = p[1], z = p[2];
  r[idx] = (bf16_t)sqrtf(x * x + y * y + z * z);
}

// ---- generic bf16 WMMA GEMM: Y[M,Nc] = A[M,K] @ Bt[Nc,K]^T (f32 out) -------
__global__ void k_gemm_bf16(const bf16_t* __restrict__ A,
                            const bf16_t* __restrict__ Bt,
                            float* __restrict__ Y, int M, int K, int Nc) {
  int wave = threadIdx.x >> 5, lane = threadIdx.x & 31;
  int l16 = lane & 15, half = lane >> 4;
  int ntn = Nc >> 4;
  int tid = blockIdx.x * 4 + wave;
  int tm = tid / ntn, tn = tid % ntn;
  const bf16_t* arow = A + (size_t)(tm * 16 + l16) * K;
  const bf16_t* brow = Bt + (size_t)(tn * 16 + l16) * K;
  v8f acc = {};
  for (int kb = 0; kb < K; kb += 32) {
    v16bf af = load_afrag(arow + kb + 16 * half);
    v16bf bf = load_bfrag(brow + kb + 16 * half);
    acc = __builtin_amdgcn_wmma_f32_16x16x32_bf16(false, af, false, bf,
                                                  (short)0, acc, false, false);
  }
#pragma unroll
  for (int r = 0; r < 8; ++r)
    Y[(size_t)(tm * 16 + r + 8 * half) * Nc + tn * 16 + l16] = acc[r];
}

// ---- LayerNorm + tanh-GELU over F=768, one wave per row --------------------
__global__ void k_ln_gelu(const float* __restrict__ Y,
                          const float* __restrict__ bias,
                          const float* __restrict__ g,
                          const float* __restrict__ be,
                          bf16_t* __restrict__ H) {
  int wave = threadIdx.x >> 5, lane = threadIdx.x & 31;
  int row = blockIdx.x * 8 + wave;
  const float* y = Y + (size_t)row * 768;
  float x[24], s = 0.f, s2 = 0.f;
#pragma unroll
  for (int t = 0; t < 24; ++t) {
    int c = lane + 32 * t;
    float v = y[c] + bias[c];
    x[t] = v; s += v; s2 += v * v;
  }
#pragma unroll
  for (int m = 16; m >= 1; m >>= 1) {
    s  += __shfl_xor(s,  m, 32);
    s2 += __shfl_xor(s2, m, 32);
  }
  float mu  = s * (1.f / 768.f);
  float var = s2 * (1.f / 768.f) - mu * mu;
  float inv = rsqrtf(var + 1e-5f);
#pragma unroll
  for (int t = 0; t < 24; ++t) {
    int c = lane + 32 * t;
    float v = (x[t] - mu) * inv * g[c] + be[c];
    float u = 0.7978845608028654f * (v + 0.044715f * v * v * v);
    H[(size_t)row * 768 + c] = (bf16_t)(0.5f * v * (1.f + tanhf(u)));
  }
}

// ---- r_dec = Y2 + bout -> bf16 ---------------------------------------------
__global__ void k_bias_cvt(const float* __restrict__ Y,
                           const float* __restrict__ bout,
                           bf16_t* __restrict__ rdec) {
  int idx = blockIdx.x * blockDim.x + threadIdx.x;   // N*C
  rdec[idx] = (bf16_t)(Y[idx] + bout[idx & 255]);
}

// ---- zero the Q x 0e tail of the output ------------------------------------
__global__ void k_zero_tail(float* __restrict__ out) {
  int idx = blockIdx.x * blockDim.x + threadIdx.x;   // N*Q
  int b = idx / 5120, q = idx - b * 5120;
  out[(size_t)b * 6656 + 1536 + q] = 0.f;
}

// ---- fused tensor-product: WMMA stage-1 + in-register stage-2 --------------
// Accumulator: M (VGPR rows) = p, N (lanes) = b.
// Each wave processes TWO b-tiles per A fragment (halves VMEM:WMMA ratio).
__global__ void k_tp(const bf16_t* __restrict__ rdec,
                     const bf16_t* __restrict__ WtpT,
                     const float* __restrict__ enc,
                     float* __restrict__ out) {
  int wave = threadIdx.x >> 5, lane = threadIdx.x & 31;
  int l16 = lane & 15, half = lane >> 4;
  int pBase = blockIdx.x * 16;                  // 32 p-tiles
  int bPair = blockIdx.y * 4 + wave;            // 32 pairs of b-tiles
  int b0 = bPair * 32 + l16;
  int b1 = b0 + 16;

  // preload B fragments (r_dec^T, K = 256) for both b-tiles: 128 VGPRs
  v16bf Bf0[8], Bf1[8];
  const bf16_t* rrow0 = rdec + (size_t)b0 * 256;
  const bf16_t* rrow1 = rdec + (size_t)b1 * 256;
#pragma unroll
  for (int c = 0; c < 8; ++c) {
    Bf0[c] = load_bfrag(rrow0 + 32 * c + 16 * half);
    Bf1[c] = load_bfrag(rrow1 + 32 * c + 16 * half);
  }

  const float* vrow0 = enc + (size_t)b0 * 768;
  const float* vrow1 = enc + (size_t)b1 * 768;
  float acc0[8][3], acc1[8][3];
#pragma unroll
  for (int r = 0; r < 8; ++r)
#pragma unroll
    for (int m = 0; m < 3; ++m) { acc0[r][m] = 0.f; acc1[r][m] = 0.f; }

  for (int j = 0; j < 256; ++j) {
    const bf16_t* ar = WtpT + ((size_t)j * 512 + pBase + l16) * 256;
    __builtin_prefetch(WtpT + ((size_t)((j + 1) & 255) * 512 + pBase + l16) * 256, 0, 1);
    v8f t0 = {}, t1 = {};
#pragma unroll
    for (int c = 0; c < 8; ++c) {
      v16bf af = load_afrag(ar + 32 * c + 16 * half);
      t0 = __builtin_amdgcn_wmma_f32_16x16x32_bf16(false, af, false, Bf0[c],
                                                   (short)0, t0, false, false);
      t1 = __builtin_amdgcn_wmma_f32_16x16x32_bf16(false, af, false, Bf1[c],
                                                   (short)0, t1, false, false);
    }
    float a0 = vrow0[j * 3 + 0], a1 = vrow0[j * 3 + 1], a2 = vrow0[j * 3 + 2];
    float c0 = vrow1[j * 3 + 0], c1 = vrow1[j * 3 + 1], c2 = vrow1[j * 3 + 2];
#pragma unroll
    for (int r = 0; r < 8; ++r) {
      acc0[r][0] += t0[r] * a0; acc0[r][1] += t0[r] * a1; acc0[r][2] += t0[r] * a2;
      acc1[r][0] += t1[r] * c0; acc1[r][1] += t1[r] * c1; acc1[r][2] += t1[r] * c2;
    }
  }

  const float sc = 1.0f / 256.0f;
#pragma unroll
  for (int r = 0; r < 8; ++r) {
    int p = pBase + r + 8 * half;
    size_t o0 = (size_t)b0 * 6656 + (size_t)p * 3;
    size_t o1 = (size_t)b1 * 6656 + (size_t)p * 3;
    out[o0 + 0] = acc0[r][0] * sc;
    out[o0 + 1] = acc0[r][1] * sc;
    out[o0 + 2] = acc0[r][2] * sc;
    out[o1 + 0] = acc1[r][0] * sc;
    out[o1 + 1] = acc1[r][1] * sc;
    out[o1 + 2] = acc1[r][2] * sc;
  }
}

// ---------------------------------------------------------------------------
extern "C" void kernel_launch(void* const* d_in, const int* in_sizes, int n_in,
                              void* d_out, int out_size, void* d_ws, size_t ws_size,
                              hipStream_t stream) {
  const float* enc  = (const float*)d_in[0];
  const float* W0   = (const float*)d_in[1];
  const float* b0   = (const float*)d_in[2];
  const float* g0   = (const float*)d_in[3];
  const float* be0  = (const float*)d_in[4];
  const float* W1   = (const float*)d_in[5];
  const float* b1   = (const float*)d_in[6];
  const float* g1   = (const float*)d_in[7];
  const float* be1  = (const float*)d_in[8];
  const float* Wout = (const float*)d_in[9];
  const float* bout = (const float*)d_in[10];
  const float* Wtp  = (const float*)d_in[11];
  float* out = (float*)d_out;

  // workspace layout (all sizes 256B-multiples); total ~76 MB
  char* w = (char*)d_ws;
  bf16_t* WtpT  = (bf16_t*)w; w += (size_t)33554432 * 2;   // [jp, i] bf16
  bf16_t* W0t   = (bf16_t*)w; w += (size_t)768 * 256 * 2;  // [F, C]
  bf16_t* W1t   = (bf16_t*)w; w += (size_t)768 * 768 * 2;  // [F, F]
  bf16_t* Woutt = (bf16_t*)w; w += (size_t)256 * 768 * 2;  // [C, F]
  bf16_t* rbf   = (bf16_t*)w; w += (size_t)1024 * 256 * 2;
  float*  Yb    = (float*)w;  w += (size_t)1024 * 768 * 4; // GEMM out buffer
  bf16_t* h0    = (bf16_t*)w; w += (size_t)1024 * 768 * 2;
  bf16_t* h1    = (bf16_t*)w; w += (size_t)1024 * 768 * 2;
  bf16_t* rdec  = (bf16_t*)w; w += (size_t)1024 * 256 * 2;

  dim3 tb(32, 8);
  // transpose+convert all weights to K-contiguous bf16
  k_transpose_cvt<<<dim3(8, 4096), tb, 0, stream>>>(Wtp,  WtpT,  256, 131072);
  k_transpose_cvt<<<dim3(8, 24),   tb, 0, stream>>>(W0,   W0t,   256, 768);
  k_transpose_cvt<<<dim3(24, 24),  tb, 0, stream>>>(W1,   W1t,   768, 768);
  k_transpose_cvt<<<dim3(24, 8),   tb, 0, stream>>>(Wout, Woutt, 768, 256);

  k_norm<<<1024, 256, 0, stream>>>(enc, rbf);

  // MLP: three WMMA GEMMs, each block = 4 waves = 4 output tiles
  k_gemm_bf16<<<768, 128, 0, stream>>>(rbf, W0t, Yb, 1024, 256, 768);
  k_ln_gelu<<<128, 256, 0, stream>>>(Yb, b0, g0, be0, h0);
  k_gemm_bf16<<<768, 128, 0, stream>>>(h0, W1t, Yb, 1024, 768, 768);
  k_ln_gelu<<<128, 256, 0, stream>>>(Yb, b1, g1, be1, h1);
  k_gemm_bf16<<<256, 128, 0, stream>>>(h1, Woutt, Yb, 1024, 768, 256);
  k_bias_cvt<<<1024, 256, 0, stream>>>(Yb, bout, rdec);

  // output: zero the structural-zero tail, then fused tensor product
  k_zero_tail<<<20480, 256, 0, stream>>>(out);
  k_tp<<<dim3(32, 8), 128, 0, stream>>>(rdec, WtpT, enc, out);
}
